// Adaptive_Attention_74483322847912
// MI455X (gfx1250) — matrix operand order repaired and checked by reading, not compiled, not run on hardware
//
#include <hip/hip_runtime.h>
#include <hip/hip_bf16.h>

typedef __bf16 bf16;
typedef __attribute__((ext_vector_type(16))) __bf16 v16bf;
typedef __attribute__((ext_vector_type(8)))  __bf16 v8bf;
typedef __attribute__((ext_vector_type(8)))  float  v8f;

__device__ __forceinline__ v8f wmma_bf16(v16bf a, v16bf b, v8f c) {
    // (neg_a, A, neg_b, B, c_mod, C, reuse_a, reuse_b)
    return __builtin_amdgcn_wmma_f32_16x16x32_bf16(false, a, false, b, (short)0, c, false, false);
}

// A fragment: 16x32 bf16, row-major source with row stride `ld` (halves).
// Lane L: M = m0 + L%16 ; lanes 0-15 hold K = 0..15, lanes 16-31 K = 16..31.
__device__ __forceinline__ v16bf load_a_frag(const bf16* s, int ld, int m0, int k0, int lane) {
    int m  = m0 + (lane & 15);
    int kb = k0 + ((lane >> 4) << 4);
    const bf16* p = s + m * ld + kb;
    v8bf lo = *(const v8bf*)p;        // K = kb .. kb+7
    v8bf hi = *(const v8bf*)(p + 8);  // K = kb+8 .. kb+15
    v16bf a;
#pragma unroll
    for (int i = 0; i < 8; ++i) { a[i] = lo[i]; a[i + 8] = hi[i]; }
    return a;
}

// B fragment from 16 contiguous halves at p (per-lane): lanes 0-15 K=0..15, 16-31 K=16..31.
__device__ __forceinline__ v16bf load_b_contig(const bf16* p) {
    v8bf lo = *(const v8bf*)p;
    v8bf hi = *(const v8bf*)(p + 8);
    v16bf b;
#pragma unroll
    for (int i = 0; i < 8; ++i) { b[i] = lo[i]; b[i + 8] = hi[i]; }
    return b;
}

__global__ void prep_weights_kernel(const float* __restrict__ w_qkv,
                                    const float* __restrict__ w_out,
                                    bf16* __restrict__ wfrag) {
    int t = blockIdx.x * blockDim.x + threadIdx.x;
    if (t >= 128 * 32) return;
    int tile = t >> 5, lane = t & 31;
    int kb = (lane >> 4) << 4;   // B frag: lanes 0-15 -> K 0..15, lanes 16-31 -> K 16..31
    int nl = lane & 15;          // N = nl within tile
    bf16* dst = wfrag + tile * 512 + lane * 16;
    if (tile < 96) {             // w_qkv tiles: [K/32=4][N/16=24]
        int kt = tile / 24, nt = tile % 24;
        const float* src = w_qkv + (kt * 32 + kb) * 384 + nt * 16 + nl;
#pragma unroll
        for (int h = 0; h < 16; ++h) dst[h] = (bf16)src[h * 384];
    } else {                     // w_out tiles: [4][8]
        int t2 = tile - 96;
        int kt = t2 >> 3, nt = t2 & 7;
        const float* src = w_out + (kt * 32 + kb) * 128 + nt * 16 + nl;
#pragma unroll
        for (int h = 0; h < 16; ++h) dst[h] = (bf16)src[h * 128];
    }
}

__global__ __launch_bounds__(256) void swin_window_attn_kernel(
    const float* __restrict__ x, const float* __restrict__ gamma,
    const float* __restrict__ beta, const float* __restrict__ bias_table,
    const bf16* __restrict__ wfrag, float* __restrict__ out) {
    // LDS: 64KB. qk[64][256] bf16 (32KB) | vt[128][64] bf16 (16KB) | 16KB overlay (xn/sim/P)
    __shared__ __align__(16) unsigned char smem[65536];
    bf16*  s_qk  = (bf16*)smem;               // q cols 0..127 | k cols 128..255 (stride 256)
    bf16*  s_vt  = (bf16*)(smem + 32768);     // V^T: [dv=128][token=64]
    bf16*  s_xn  = (bf16*)(smem + 49152);     // [64][128] bf16 (phase 1)
    float* s_sim = (float*)(smem + 49152);    // [64][64]  f32  (attention)
    bf16*  s_p   = (bf16*)(smem + 49152);     // P rows at 128-half stride (in-place)

    const int tid = threadIdx.x, lane = tid & 31, wave = tid >> 5;
    const long win = blockIdx.x;
    const float* xw = x + win * (64 * 128);
    const float scale = 0.17677669529663687f;  // dh^-0.5, dh=32
    v8f zero = {};

    // Prefetch this block's 32KB x window (global_prefetch_b8): 256 thr x 128B
    __builtin_prefetch(xw + tid * 32, 0, 0);

    // ---- 1. LayerNorm over d=128, fp32, write bf16 xn ----
    float4 gv = ((const float4*)gamma)[lane];
    float4 bv = ((const float4*)beta)[lane];
    for (int row = wave * 8; row < wave * 8 + 8; ++row) {
        float4 v = ((const float4*)(xw + row * 128))[lane];
        float s  = v.x + v.y + v.z + v.w;
        float s2 = v.x * v.x + v.y * v.y + v.z * v.z + v.w * v.w;
#pragma unroll
        for (int off = 16; off > 0; off >>= 1) {
            s  += __shfl_xor(s,  off, 32);
            s2 += __shfl_xor(s2, off, 32);
        }
        float mu   = s * (1.0f / 128.0f);
        float var  = s2 * (1.0f / 128.0f) - mu * mu;
        float rstd = rsqrtf(var + 1e-5f);
        bf16* dst = s_xn + row * 128 + lane * 4;
        dst[0] = (bf16)((v.x - mu) * rstd * gv.x + bv.x);
        dst[1] = (bf16)((v.y - mu) * rstd * gv.y + bv.y);
        dst[2] = (bf16)((v.z - mu) * rstd * gv.z + bv.z);
        dst[3] = (bf16)((v.w - mu) * rstd * gv.w + bv.w);
    }
    __syncthreads();

    // ---- 2. QKV = xn @ w_qkv : 4x24 tiles, K=128 in 4 steps ----
    for (int t = wave; t < 96; t += 8) {
        int mt = t / 24, nt = t % 24;
        v8f acc = zero;
#pragma unroll
        for (int kk = 0; kk < 4; ++kk) {
            v16bf a = load_a_frag(s_xn, 128, mt * 16, kk * 32, lane);
            v16bf b = *(const v16bf*)(wfrag + (kk * 24 + nt) * 512 + lane * 16);
            acc = wmma_bf16(a, b, acc);
        }
        int col   = nt * 16 + (lane & 15);          // 0..127 Q | 128..255 K | 256..383 V
        int mbase = mt * 16 + ((lane >> 4) << 3);
        if (nt < 16) {
            float sc = (nt < 8) ? scale : 1.0f;     // fold softmax scale into Q
#pragma unroll
            for (int r = 0; r < 8; ++r)
                s_qk[(mbase + r) * 256 + col] = (bf16)(acc[r] * sc);
        } else {
            // V stored transposed: 8 accumulator rows -> 8 contiguous halves (one b128 store)
            int dv = col - 256;
            v8bf pv;
#pragma unroll
            for (int r = 0; r < 8; ++r) pv[r] = (bf16)acc[r];
            *(v8bf*)(s_vt + dv * 64 + mbase) = pv;
        }
    }
    __syncthreads();

    // ---- 3. Attention, one head at a time (dh = 32 -> single WMMA K step) ----
    for (int h = 0; h < 4; ++h) {
        // sim = Q_h @ K_h^T : 4x4 tiles, 2 per wave
        for (int t = wave * 2; t < wave * 2 + 2; ++t) {
            int mt = t >> 2, nt = t & 3;
            v16bf a = load_a_frag(s_qk, 256, mt * 16, h * 32, lane);
            // B[k][n] = K_h[n][k]: 16 contiguous halves along token-row of K
            v16bf b = load_b_contig(s_qk + (nt * 16 + (lane & 15)) * 256 + 128 + h * 32
                                    + ((lane >> 4) << 4));
            v8f acc = wmma_bf16(a, b, zero);
            int j  = nt * 16 + (lane & 15);
            int jh = j >> 3, jw = j & 7;
            int ibase = mt * 16 + ((lane >> 4) << 3);
#pragma unroll
            for (int r = 0; r < 8; ++r) {
                int i   = ibase + r;
                int rel = ((i >> 3) - jh + 7) * 15 + ((i & 7) - jw + 7);
                s_sim[i * 64 + j] = acc[r] + bias_table[rel * 4 + h];
            }
        }
        __syncthreads();

        // softmax over 64 cols; 4 lanes/row; in-place fp32 -> bf16 P
        {
            int row = tid >> 2, quad = tid & 3;
            const float* sr = s_sim + row * 64 + quad * 16;
            float vals[16];
            float mx = -3.4e38f;
#pragma unroll
            for (int c = 0; c < 16; ++c) { vals[c] = sr[c]; mx = fmaxf(mx, vals[c]); }
            mx = fmaxf(mx, __shfl_xor(mx, 1, 32));
            mx = fmaxf(mx, __shfl_xor(mx, 2, 32));
            float sum = 0.0f;
#pragma unroll
            for (int c = 0; c < 16; ++c) { vals[c] = __expf(vals[c] - mx); sum += vals[c]; }
            sum += __shfl_xor(sum, 1, 32);
            sum += __shfl_xor(sum, 2, 32);
            float inv = 1.0f / sum;
            bf16* pw = s_p + row * 128 + quad * 16;  // P row at byte offset row*256
#pragma unroll
            for (int c = 0; c < 16; ++c) pw[c] = (bf16)(vals[c] * inv);
        }
        __syncthreads();

        // out_h = P @ V_h : 4x2 tiles, 1 per wave; write over dead Q_h columns
        {
            int mt = wave >> 1, ntv = wave & 1;
            v8f acc = zero;
#pragma unroll
            for (int kk = 0; kk < 2; ++kk) {
                v16bf a = load_a_frag(s_p, 128, mt * 16, kk * 32, lane);
                // B[k=token][n=dv] from V^T: contiguous halves along a V^T row
                int dv = h * 32 + ntv * 16 + (lane & 15);
                v16bf b = load_b_contig(s_vt + dv * 64 + kk * 32 + ((lane >> 4) << 4));
                acc = wmma_bf16(a, b, acc);
            }
            int c = h * 32 + ntv * 16 + (lane & 15);
            int ibase = mt * 16 + ((lane >> 4) << 3);
#pragma unroll
            for (int r = 0; r < 8; ++r)
                s_qk[(ibase + r) * 256 + c] = (bf16)acc[r];
        }
        __syncthreads();
    }

    // ---- 4. out = attn_out @ w_out : 4x8 tiles, K=128 in 4 steps ----
    float* ow = out + win * (64 * 128);
    for (int t = wave; t < 32; t += 8) {
        int mt = t >> 3, nt = t & 7;
        v8f acc = zero;
#pragma unroll
        for (int kk = 0; kk < 4; ++kk) {
            v16bf a = load_a_frag(s_qk, 256, mt * 16, kk * 32, lane);
            v16bf b = *(const v16bf*)(wfrag + (96 + kk * 8 + nt) * 512 + lane * 16);
            acc = wmma_bf16(a, b, acc);
        }
        int col   = nt * 16 + (lane & 15);
        int ibase = mt * 16 + ((lane >> 4) << 3);
#pragma unroll
        for (int r = 0; r < 8; ++r)
            ow[(ibase + r) * 128 + col] = acc[r];
    }
}

extern "C" void kernel_launch(void* const* d_in, const int* in_sizes, int n_in,
                              void* d_out, int out_size, void* d_ws, size_t ws_size,
                              hipStream_t stream) {
    const float* x          = (const float*)d_in[0];
    const float* gamma      = (const float*)d_in[1];
    const float* beta       = (const float*)d_in[2];
    const float* w_qkv      = (const float*)d_in[3];
    const float* w_out      = (const float*)d_in[4];
    const float* bias_table = (const float*)d_in[5];
    // d_in[6], d_in[7]: window_height/width = 8 (hard-coded)
    bf16* wfrag = (bf16*)d_ws;  // 128 tiles * 512 bf16 = 128 KB

    prep_weights_kernel<<<16, 256, 0, stream>>>(w_qkv, w_out, wfrag);
    swin_window_attn_kernel<<<4096, 256, 0, stream>>>(x, gamma, beta, bias_table,
                                                      wfrag, (float*)d_out);
}
